// HyperdimensionalConsciousnessNetwork_77257871721086
// MI455X (gfx1250) — matrix operand order repaired and checked by reading, not compiled, not run on hardware
//
#include <hip/hip_runtime.h>

// ---------------------------------------------------------------------------
// MI455X (gfx1250) implementation.
// Compute-bound problem (~180 GFLOP vs ~150MB HBM traffic) -> drive it through
// v_wmma_f32_16x16x32_bf16 with fp32 accumulation. Fragments are loaded with
// global b128 loads in contiguous k order: lane half h holds k = 16h..16h+15.
// Round 2: 64x64 wave tile (4x4 16x16 accumulators, 16 wmma / 8KB fragment
// loads per K=32 step) to double register-level arithmetic intensity.
// ---------------------------------------------------------------------------

typedef __attribute__((ext_vector_type(16))) __bf16 v16bf;
typedef __attribute__((ext_vector_type(8)))  __bf16 v8bf;
typedef __attribute__((ext_vector_type(8)))  float  v8f;

__device__ __forceinline__ unsigned short f2bf(float f) {
  // round-to-nearest-even f32 -> bf16
  unsigned int u = __float_as_uint(f);
  u += 0x7FFFu + ((u >> 16) & 1u);
  return (unsigned short)(u >> 16);
}

__device__ __forceinline__ v16bf load_frag16(const unsigned short* p0,
                                             const unsigned short* p1) {
  v8bf lo = *(const v8bf*)p0;   // global_load_b128
  v8bf hi = *(const v8bf*)p1;   // global_load_b128
  return __builtin_shufflevector(lo, hi, 0, 1, 2, 3, 4, 5, 6, 7,
                                         8, 9, 10, 11, 12, 13, 14, 15);
}

// --------------------------- Vesica + bf16 convert --------------------------
// One block per row: reduce ||x[b]-x[b-1]||, then emit bf16 of x*(1+overlap).
__global__ __launch_bounds__(256) void vesica_scale_cvt(
    const float* __restrict__ x, unsigned short* __restrict__ xb,
    int Brows, int D) {
  const int b  = blockIdx.x;
  const int t  = threadIdx.x;
  const int bm = (b == 0) ? (Brows - 1) : (b - 1);
  const float* xr = x + (size_t)b * D;
  const float* xp = x + (size_t)bm * D;

  float s = 0.f;
  for (int d = t; d < D; d += 256) {
    float df = xr[d] - xp[d];
    s += df * df;
  }
  __shared__ float red[256];
  red[t] = s;
  __syncthreads();
  for (int w = 128; w > 0; w >>= 1) {
    if (t < w) red[t] += red[t + w];
    __syncthreads();
  }
  const float dist    = sqrtf(red[0]);
  const float overlap = fmaxf(0.f, 1.f - fabsf(dist - 1.f));
  const float scale   = 1.f + overlap;

  unsigned short* xo = xb + (size_t)b * D;
  for (int d = t; d < D; d += 256) xo[d] = f2bf(xr[d] * scale);
}

// --------------------------- weight bf16 converts ---------------------------
__global__ __launch_bounds__(256) void cvt_bf16(
    const float* __restrict__ in, unsigned short* __restrict__ o, int n) {
  int i = blockIdx.x * 256 + threadIdx.x;
  if (i < n) o[i] = f2bf(in[i]);
}

// female path: fold the x_expanded[:, ::-1] onto the weights (reverse columns)
__global__ __launch_bounds__(256) void cvt_bf16_revcols(
    const float* __restrict__ in, unsigned short* __restrict__ o,
    int rows, int cols) {
  int i = blockIdx.x * 256 + threadIdx.x;
  if (i < rows * cols) {
    int r = i / cols, c = i % cols;
    o[(size_t)r * cols + c] = f2bf(in[(size_t)r * cols + (cols - 1 - c)]);
  }
}

// ------------------------------- WMMA GEMM ---------------------------------
// C[M,N] = A[M,K] * W[N,K]^T.  A,W bf16 row-major.
// Block: 8 waves (2 in M x 4 in N), wave tile 64x64 -> block tile 128x256.
// PHASE_EPI=true : scale columns h with h%512<2 by cos/sin(2*pi*(h/512)/7),
//                  store bf16 (this is x_expanded for the Merkaba GEMMs).
// PHASE_EPI=false: store fp32.
template <bool PHASE_EPI>
__global__ __launch_bounds__(256, 1) void gemm_wmma_bf16(
    const unsigned short* __restrict__ A, const unsigned short* __restrict__ W,
    void* __restrict__ Cout, int K, int N) {
  const int tid   = threadIdx.x;
  const int wave  = tid >> 5;
  const int lane  = tid & 31;
  const int waveM = wave & 1;
  const int waveN = wave >> 1;
  const int M0    = blockIdx.x * 128 + waveM * 64;
  const int N0    = blockIdx.y * 256 + waveN * 64;
  const int m     = lane & 15;   // row (A) / col (B) within 16-tile
  const int kh    = lane >> 4;   // K-half select per CDNA5 layout

  // A fragment (16x32, 16-bit): lane holds row m, K-range kh*16..+15.
  const unsigned short* aP[4];
#pragma unroll
  for (int i = 0; i < 4; ++i)
    aP[i] = A + (size_t)(M0 + 16 * i + m) * K + kh * 16;

  // B fragment (32x16, 16-bit): lane holds col m, K-range kh*16..+15,
  // contiguous in W's row-major [N,K] storage.
  const unsigned short* bP[4];
#pragma unroll
  for (int j = 0; j < 4; ++j)
    bP[j] = W + (size_t)(N0 + 16 * j + m) * K + kh * 16;

  v8f acc[4][4] = {};

  for (int k0 = 0; k0 < K; k0 += 32) {
    v16bf a[4], b[4];
#pragma unroll
    for (int i = 0; i < 4; ++i)
      a[i] = load_frag16(aP[i] + k0, aP[i] + k0 + 8);
#pragma unroll
    for (int j = 0; j < 4; ++j)
      b[j] = load_frag16(bP[j] + k0, bP[j] + k0 + 8);
#pragma unroll
    for (int i = 0; i < 4; ++i)
#pragma unroll
      for (int j = 0; j < 4; ++j)
        acc[i][j] = __builtin_amdgcn_wmma_f32_16x16x32_bf16(
            false, a[i], false, b[j], (short)0, acc[i][j], false, false);
  }

  // C/D layout: VGPR e -> row = 16*i + kh*8 + e, col = 16*j + m.
  const int rbase = kh * 8;
#pragma unroll
  for (int i = 0; i < 4; ++i) {
#pragma unroll
    for (int j = 0; j < 4; ++j) {
      const int col = N0 + 16 * j + m;
      float mul = 1.0f;
      if (PHASE_EPI) {
        const int r = col & 511;
        if (r < 2) {
          const float ph = 0.8975979010256552f * (float)(col >> 9); // 2*pi/7*c
          mul = (r == 0) ? __cosf(ph) : __sinf(ph);
        }
      }
#pragma unroll
      for (int e = 0; e < 8; ++e) {
        const int row = M0 + 16 * i + rbase + e;
        const float v = acc[i][j][e] * mul;
        if (PHASE_EPI)
          ((unsigned short*)Cout)[(size_t)row * N + col] = f2bf(v);
        else
          ((float*)Cout)[(size_t)row * N + col] = v;
      }
    }
  }
}

// ------------------------------ Merkaba blend ------------------------------
__global__ __launch_bounds__(256) void merkaba_blend(
    const float* __restrict__ male, const float* __restrict__ female,
    float* __restrict__ out, int D) {
  const int row = blockIdx.x;
  const int t   = threadIdx.x;
  const float* mr = male + (size_t)row * D;
  const float* fr = female + (size_t)row * D;

  float s = 0.f;
  for (int d = t; d < D; d += 256) s += mr[d] * fr[d];
  __shared__ float red[256];
  red[t] = s;
  __syncthreads();
  for (int w = 128; w > 0; w >>= 1) {
    if (t < w) red[t] += red[t + w];
    __syncthreads();
  }
  const float p = 1.f / (1.f + __expf(-red[0]));
  float* orow = out + (size_t)row * D;
  for (int d = t; d < D; d += 256)
    orow[d] = p * mr[d] + (1.f - p) * fr[d];
}

// -------------------------------- launcher ---------------------------------
extern "C" void kernel_launch(void* const* d_in, const int* in_sizes, int n_in,
                              void* d_out, int out_size, void* d_ws,
                              size_t ws_size, hipStream_t stream) {
  (void)in_sizes; (void)n_in; (void)out_size; (void)ws_size;

  constexpr int Bm = 4096, DIN = 2048, DHID = 3584, DOUT = 2048;

  const float* x     = (const float*)d_in[0];
  const float* Wfl   = (const float*)d_in[1];  // [7,512,2048] == [3584,2048]
  const float* Wmale = (const float*)d_in[2];  // [2048,3584]
  const float* Wfem  = (const float*)d_in[3];  // [2048,3584]
  float* out = (float*)d_out;

  // Workspace carving (256B aligned). Total ~157 MB.
  char* ws = (char*)d_ws;
  size_t off = 0;
  auto carve = [&](size_t bytes) -> void* {
    void* p = ws + off;
    off += (bytes + 255) & ~(size_t)255;
    return p;
  };
  unsigned short* xbf    = (unsigned short*)carve((size_t)Bm * DIN * 2);
  unsigned short* wflbf  = (unsigned short*)carve((size_t)DHID * DIN * 2);
  unsigned short* wmbf   = (unsigned short*)carve((size_t)DOUT * DHID * 2);
  unsigned short* wfrbf  = (unsigned short*)carve((size_t)DOUT * DHID * 2);
  unsigned short* xexpbf = (unsigned short*)carve((size_t)Bm * DHID * 2);
  float* male   = (float*)carve((size_t)Bm * DOUT * 4);
  float* female = (float*)carve((size_t)Bm * DOUT * 4);

  // 1) Vesica scale + bf16 conversion of x_overlap
  vesica_scale_cvt<<<Bm, 256, 0, stream>>>(x, xbf, Bm, DIN);

  // 2) bf16 weight conversions (female with reversed columns)
  {
    int n = DHID * DIN;
    cvt_bf16<<<(n + 255) / 256, 256, 0, stream>>>(Wfl, wflbf, n);
    n = DOUT * DHID;
    cvt_bf16<<<(n + 255) / 256, 256, 0, stream>>>(Wmale, wmbf, n);
    cvt_bf16_revcols<<<(n + 255) / 256, 256, 0, stream>>>(Wfem, wfrbf, DOUT,
                                                          DHID);
  }

  // 3) Flower GEMM: [4096,2048] x [3584,2048]^T -> bf16 x_expanded (+phase)
  gemm_wmma_bf16<true><<<dim3(Bm / 128, DHID / 256), 256, 0, stream>>>(
      xbf, wflbf, (void*)xexpbf, DIN, DHID);

  // 4) Merkaba GEMMs: male / female, fp32 outputs
  gemm_wmma_bf16<false><<<dim3(Bm / 128, DOUT / 256), 256, 0, stream>>>(
      xexpbf, wmbf, (void*)male, DHID, DOUT);
  gemm_wmma_bf16<false><<<dim3(Bm / 128, DOUT / 256), 256, 0, stream>>>(
      xexpbf, wfrbf, (void*)female, DHID, DOUT);

  // 5) Sigmoid phase lock + blend
  merkaba_blend<<<Bm, 256, 0, stream>>>(male, female, out, DOUT);
}
